// MambaClassifier_78546361909272
// MI455X (gfx1250) — matrix operand order repaired and checked by reading, not compiled, not run on hardware
//
#include <hip/hip_runtime.h>
#include <hip/hip_bf16.h>
#include <math.h>
#include <stdint.h>

#define B_SZ     16
#define L_SZ     2048
#define D_INPUT  64
#define DM       512
#define DSTATE   16
#define NLAYERS  4
#define M_TOTAL  (B_SZ * L_SZ)   // 32768 rows

typedef __attribute__((ext_vector_type(16))) __bf16 v16bf;
typedef __attribute__((ext_vector_type(8)))  float  v8f;

#define LDA 40   // LDS row stride (halves) for A tile [128 x 32]; 80B = 16B aligned
#define LDB 40   // LDS row stride (halves) for Wt tile [64 x 32]

__device__ __forceinline__ float softplus_f(float x) {
    return (x > 20.f) ? x : __logf(1.f + __expf(x));
}

// f32 -> bf16 elementwise
__global__ __launch_bounds__(256)
void cvt_f32_bf16(const float* __restrict__ in, __bf16* __restrict__ out, int n) {
    int i = blockIdx.x * 256 + threadIdx.x;
    if (i < n) out[i] = (__bf16)in[i];
}

// W (KxN, f32) -> Wt (NxK, bf16)
__global__ __launch_bounds__(256)
void transpose_cvt(const float* __restrict__ W, __bf16* __restrict__ Wt, int K, int N) {
    int e = blockIdx.x * 256 + threadIdx.x;
    if (e < K * N) {
        int k = e / N, n = e % N;
        Wt[(size_t)n * K + k] = (__bf16)W[e];
    }
}

// out = act( A(MxK,bf16) @ Wt^T(NxK,bf16) + bias ), f32 accumulate via
// v_wmma_f32_16x16x32_bf16. Tiles staged to LDS with double-buffered
// global_load_async_to_lds_b128 (ASYNCcnt), 3 async instrs per wave per tile.
// grid = (M/128, ceil(N/64)), block = 256 (8 wave32 waves -> 32x32 subtiles).
__global__ __launch_bounds__(256)
void gemm_bf16_wmma_async(const __bf16* __restrict__ A, const __bf16* __restrict__ Wt,
                          const float* __restrict__ bias,
                          float* __restrict__ outF, __bf16* __restrict__ outH,
                          int N, int K, int act) {
    __shared__ __bf16 As[2][128 * LDA];   // [row][k]
    __shared__ __bf16 Bs[2][64 * LDB];    // [n][k]

    const int tid  = threadIdx.x;
    const int lane = tid & 31;
    const int wave = tid >> 5;
    const int wm   = wave & 3;
    const int wn   = wave >> 2;
    const int half = lane >> 4;
    const int lr   = lane & 15;

    const int m0 = blockIdx.x * 128;
    const int n0 = blockIdx.y * 64;

    // per-thread 16B async-copy chunks: A tile 128x32 halves = 512 chunks (2/thread),
    // Wt tile 64x32 halves = 256 chunks (1/thread)
    const int cA1   = tid + 256;
    const int rowA0 = tid >> 2,  kqA0 = (tid & 3) << 3;
    const int rowA1 = cA1 >> 2,  kqA1 = (cA1 & 3) << 3;
    const int rowW  = tid >> 2,  kqW  = (tid & 3) << 3;
    const int rowWc = (n0 + rowW < N) ? (n0 + rowW) : (N - 1);  // clamp, never branch

    auto issue_tile = [&](int buf, int ktile) {
        const __bf16* a0 = A  + (size_t)(m0 + rowA0) * K + ktile * 32 + kqA0;
        const __bf16* a1 = A  + (size_t)(m0 + rowA1) * K + ktile * 32 + kqA1;
        const __bf16* w  = Wt + (size_t)rowWc        * K + ktile * 32 + kqW;
        uint32_t l0 = (uint32_t)(uintptr_t)&As[buf][rowA0 * LDA + kqA0];
        uint32_t l1 = (uint32_t)(uintptr_t)&As[buf][rowA1 * LDA + kqA1];
        uint32_t lw = (uint32_t)(uintptr_t)&Bs[buf][rowW  * LDB + kqW];
        asm volatile("global_load_async_to_lds_b128 %0, %1, off" :: "v"(l0), "v"(a0) : "memory");
        asm volatile("global_load_async_to_lds_b128 %0, %1, off" :: "v"(l1), "v"(a1) : "memory");
        asm volatile("global_load_async_to_lds_b128 %0, %1, off" :: "v"(lw), "v"(w)  : "memory");
    };

    v8f acc[2][2];
    #pragma unroll
    for (int i = 0; i < 2; ++i)
        #pragma unroll
        for (int j = 0; j < 2; ++j) { v8f z = {}; acc[i][j] = z; }

    const int T = K >> 5;
    issue_tile(0, 0);

    for (int kt = 0; kt < T; ++kt) {
        if (kt + 1 < T) {
            issue_tile((kt + 1) & 1, kt + 1);
            asm volatile("s_wait_asynccnt 0x3" ::: "memory");  // prev tile (3 instrs) done
        } else {
            asm volatile("s_wait_asynccnt 0x0" ::: "memory");
        }
        __syncthreads();

        const __bf16* Asb = As[kt & 1];
        const __bf16* Bsb = Bs[kt & 1];

        // fragment layouts used here (wave32, contiguous k, same order for A and B):
        v16bf af[2], bfv[2];
        #pragma unroll
        for (int i = 0; i < 2; ++i) {
            // A 16x32: lanes 0-15 -> K 0..15 ; lanes 16-31 -> K 16..31
            const __bf16* p = &Asb[(wm * 32 + i * 16 + lr) * LDA + half * 16];
            #pragma unroll
            for (int q = 0; q < 8; ++q) { af[i][q] = p[q]; af[i][8 + q] = p[8 + q]; }
        }
        #pragma unroll
        for (int j = 0; j < 2; ++j) {
            // B 32x16: n = lane%16; lanes 0-15 hold K=0..15, lanes 16-31 K=16..31
            const __bf16* p = &Bsb[(wn * 32 + j * 16 + lr) * LDB + half * 16];
            #pragma unroll
            for (int q = 0; q < 16; ++q) bfv[j][q] = p[q];
        }
        #pragma unroll
        for (int i = 0; i < 2; ++i)
            #pragma unroll
            for (int j = 0; j < 2; ++j)
                acc[i][j] = __builtin_amdgcn_wmma_f32_16x16x32_bf16(
                    false, af[i], false, bfv[j], (short)0, acc[i][j], false, false);
        __syncthreads();   // tile buf reused two iterations later
    }

    #pragma unroll
    for (int i = 0; i < 2; ++i)
        #pragma unroll
        for (int j = 0; j < 2; ++j)
            #pragma unroll
            for (int r = 0; r < 8; ++r) {
                int row = m0 + wm * 32 + i * 16 + r + (half ? 8 : 0);
                int col = n0 + wn * 32 + j * 16 + lr;
                if (col < N) {
                    float v = acc[i][j][r];
                    if (bias) v += bias[col];
                    if (act == 1) v = softplus_f(v);
                    if (outF) outF[(size_t)row * N + col] = v;
                    if (outH) outH[(size_t)row * N + col] = (__bf16)v;
                }
            }
}

// Selective scan: one thread per channel, h[16] state in registers, serial over L.
// Writes bf16 y for the next layer's GEMM; f32 only for the last token (LN input).
// grid = (B, DM/256), block = 256.
__global__ __launch_bounds__(256)
void mamba_scan(const float* __restrict__ dt, const float* __restrict__ u,
                const float* __restrict__ Bm, const float* __restrict__ Cm,
                const float* __restrict__ A_log, const float* __restrict__ Dskip,
                __bf16* __restrict__ yh, float* __restrict__ ylast) {
    const int b = blockIdx.x;
    const int d = blockIdx.y * 256 + threadIdx.x;

    float Arow[DSTATE];
    #pragma unroll
    for (int n = 0; n < DSTATE; ++n) Arow[n] = -__expf(A_log[d * DSTATE + n]);
    const float Dv = Dskip[d];

    float h[DSTATE];
    #pragma unroll
    for (int n = 0; n < DSTATE; ++n) h[n] = 0.f;

    const size_t base = (size_t)b * L_SZ;
    for (int t = 0; t < L_SZ; ++t) {
        const size_t rt = base + t;
        const float dtv = dt[rt * DM + d];
        const float uv  = u[rt * DM + d];
        const float4* Bp = reinterpret_cast<const float4*>(Bm + rt * DSTATE);
        const float4* Cp = reinterpret_cast<const float4*>(Cm + rt * DSTATE);
        float Bv[DSTATE], Cv[DSTATE];
        #pragma unroll
        for (int q = 0; q < 4; ++q) {
            float4 bv = Bp[q], cv = Cp[q];
            Bv[q*4+0] = bv.x; Bv[q*4+1] = bv.y; Bv[q*4+2] = bv.z; Bv[q*4+3] = bv.w;
            Cv[q*4+0] = cv.x; Cv[q*4+1] = cv.y; Cv[q*4+2] = cv.z; Cv[q*4+3] = cv.w;
        }
        const float du = dtv * uv;
        float accv = 0.f;
        #pragma unroll
        for (int n = 0; n < DSTATE; ++n) {
            h[n] = __expf(dtv * Arow[n]) * h[n] + du * Bv[n];
            accv += h[n] * Cv[n];
        }
        const float yv = accv + uv * Dv;
        yh[rt * DM + d] = (__bf16)yv;
        if (t == L_SZ - 1) ylast[(size_t)b * DM + d] = yv;
    }
}

// LayerNorm of last token + head. grid = B, block = DM.
__global__ __launch_bounds__(512)
void ln_head(const float* __restrict__ ylast, const float* __restrict__ g,
             const float* __restrict__ be, const float* __restrict__ hw,
             const float* __restrict__ hb, float* __restrict__ out) {
    __shared__ float red[DM];
    const int b = blockIdx.x, d = threadIdx.x;
    float v = ylast[(size_t)b * DM + d];

    red[d] = v; __syncthreads();
    for (int s = DM / 2; s > 0; s >>= 1) { if (d < s) red[d] += red[d + s]; __syncthreads(); }
    const float mu = red[0] / (float)DM; __syncthreads();

    const float c = v - mu;
    red[d] = c * c; __syncthreads();
    for (int s = DM / 2; s > 0; s >>= 1) { if (d < s) red[d] += red[d + s]; __syncthreads(); }
    const float var = red[0] / (float)DM; __syncthreads();

    const float nv = c * rsqrtf(var + 1e-5f) * g[d] + be[d];
    red[d] = nv * hw[d]; __syncthreads();
    for (int s = DM / 2; s > 0; s >>= 1) { if (d < s) red[d] += red[d + s]; __syncthreads(); }
    if (d == 0) out[b] = red[0] + hb[0];
}

extern "C" void kernel_launch(void* const* d_in, const int* in_sizes, int n_in,
                              void* d_out, int out_size, void* d_ws, size_t ws_size,
                              hipStream_t stream) {
    const float* x      = (const float*)d_in[0];
    const float* W_in0  = (const float*)d_in[1];
    const float* b_in0  = (const float*)d_in[2];
    const float* W_in   = (const float*)d_in[3];
    const float* b_in   = (const float*)d_in[4];
    const float* A_log  = (const float*)d_in[5];
    const float* Dskip  = (const float*)d_in[6];
    const float* W_dt   = (const float*)d_in[7];
    const float* b_dt   = (const float*)d_in[8];
    const float* W_B    = (const float*)d_in[9];
    const float* W_C    = (const float*)d_in[10];
    const float* ln_g   = (const float*)d_in[11];
    const float* ln_b   = (const float*)d_in[12];
    const float* head_w = (const float*)d_in[13];
    const float* head_b = (const float*)d_in[14];
    float* out = (float*)d_out;

    char* ws = (char*)d_ws;
    const size_t szBLD  = (size_t)M_TOTAL * DM * sizeof(float);        // 64 MB
    const size_t szBLN  = (size_t)M_TOTAL * DSTATE * sizeof(float);    // 2 MB
    const size_t szBLDh = (size_t)M_TOTAL * DM * sizeof(__bf16);       // 32 MB
    const size_t szXh   = (size_t)M_TOTAL * D_INPUT * sizeof(__bf16);  // 4 MB

    size_t off = 0;
    float*  u    = (float*)(ws + off);  off += szBLD;
    float*  dt   = (float*)(ws + off);  off += szBLD;
    float*  Bm   = (float*)(ws + off);  off += szBLN;
    float*  Cm   = (float*)(ws + off);  off += szBLN;
    float*  ylst = (float*)(ws + off);  off += (size_t)B_SZ * DM * sizeof(float);
    off = (off + 255) & ~(size_t)255;
    __bf16* xb   = (__bf16*)(ws + off); off += szXh;
    __bf16* ub   = (__bf16*)(ws + off); off += szBLDh;
    __bf16* yb   = (__bf16*)(ws + off); off += szBLDh;
    __bf16* w0t  = (__bf16*)(ws + off); off += (size_t)DM * D_INPUT * 2;       // 512x64
    __bf16* wint = (__bf16*)(ws + off); off += (size_t)(NLAYERS-1) * DM * DM * 2;
    __bf16* wdtt = (__bf16*)(ws + off); off += (size_t)NLAYERS * DM * DM * 2;
    __bf16* wBt  = (__bf16*)(ws + off); off += (size_t)NLAYERS * DSTATE * DM * 2;
    __bf16* wCt  = (__bf16*)(ws + off); off += (size_t)NLAYERS * DSTATE * DM * 2;

    dim3 blk(256);

    // ---- one-time (per launch) bf16 conversion / weight transposition ----
    cvt_f32_bf16<<<dim3((M_TOTAL * D_INPUT + 255) / 256), blk, 0, stream>>>(x, xb, M_TOTAL * D_INPUT);
    transpose_cvt<<<dim3((D_INPUT * DM + 255) / 256), blk, 0, stream>>>(W_in0, w0t, D_INPUT, DM);
    for (int l = 0; l < NLAYERS - 1; ++l)
        transpose_cvt<<<dim3((DM * DM + 255) / 256), blk, 0, stream>>>(
            W_in + (size_t)l * DM * DM, wint + (size_t)l * DM * DM, DM, DM);
    for (int l = 0; l < NLAYERS; ++l) {
        transpose_cvt<<<dim3((DM * DM + 255) / 256), blk, 0, stream>>>(
            W_dt + (size_t)l * DM * DM, wdtt + (size_t)l * DM * DM, DM, DM);
        transpose_cvt<<<dim3((DM * DSTATE + 255) / 256), blk, 0, stream>>>(
            W_B + (size_t)l * DM * DSTATE, wBt + (size_t)l * DSTATE * DM, DM, DSTATE);
        transpose_cvt<<<dim3((DM * DSTATE + 255) / 256), blk, 0, stream>>>(
            W_C + (size_t)l * DM * DSTATE, wCt + (size_t)l * DSTATE * DM, DM, DSTATE);
    }

    const dim3 gFull(M_TOTAL / 128, DM / 64);
    const dim3 gSmall(M_TOTAL / 128, 1);
    const dim3 gScan(B_SZ, DM / 256);

    for (int l = 0; l < NLAYERS; ++l) {
        const __bf16* Ain = (l == 0) ? xb : yb;
        const int     K   = (l == 0) ? D_INPUT : DM;
        const __bf16* Wt  = (l == 0) ? w0t : (wint + (size_t)(l - 1) * DM * DM);
        const float*  bi  = (l == 0) ? b_in0 : (b_in + (size_t)(l - 1) * DM);

        gemm_bf16_wmma_async<<<gFull, blk, 0, stream>>>(Ain, Wt, bi, u, ub, DM, K, 0);
        gemm_bf16_wmma_async<<<gFull, blk, 0, stream>>>(ub, wdtt + (size_t)l * DM * DM,
                                                        b_dt + (size_t)l * DM, dt, nullptr, DM, DM, 1);
        gemm_bf16_wmma_async<<<gSmall, blk, 0, stream>>>(ub, wBt + (size_t)l * DSTATE * DM,
                                                         nullptr, Bm, nullptr, DSTATE, DM, 0);
        gemm_bf16_wmma_async<<<gSmall, blk, 0, stream>>>(ub, wCt + (size_t)l * DSTATE * DM,
                                                         nullptr, Cm, nullptr, DSTATE, DM, 0);
        mamba_scan<<<gScan, blk, 0, stream>>>(dt, u, Bm, Cm,
                                              A_log + (size_t)l * DM * DSTATE,
                                              Dskip + (size_t)l * DM, yb, ylst);
    }
    ln_head<<<dim3(B_SZ), dim3(DM), 0, stream>>>(ylst, ln_g, ln_b, head_w, head_b, out);
}
